// SlidingWindowAttention_83339545411988
// MI455X (gfx1250) — matrix operand order repaired and checked by reading, not compiled, not run on hardware
//
#include <hip/hip_runtime.h>
#include <hip/hip_bf16.h>
#include <stdint.h>

// ---------------------------------------------------------------------------
// CDNA5 wave32 WMMA types
// ---------------------------------------------------------------------------
typedef __attribute__((ext_vector_type(16))) __bf16        v16bf;
typedef __attribute__((ext_vector_type(8)))  float         v8f;

struct U16x16 { uint4 lo, hi; };

static __device__ __forceinline__ unsigned short f2bf(float f) {
  union { float f; unsigned int u; } v; v.f = f;
  unsigned int u = v.u;
  return (unsigned short)((u + 0x7FFFu + ((u >> 16) & 1u)) >> 16);
}

// Build a 16-element bf16 fragment from two contiguous 8-element (16B) chunks.
static __device__ __forceinline__ v16bf make_frag(const unsigned short* p0,
                                                  const unsigned short* p1) {
  U16x16 t;
  t.lo = *(const uint4*)p0;
  t.hi = *(const uint4*)p1;
  return __builtin_bit_cast(v16bf, t);
}

static __device__ __forceinline__ v8f wmma_bf16(v16bf a, v16bf b, v8f c) {
  return __builtin_amdgcn_wmma_f32_16x16x32_bf16(
      /*neg_a=*/false, a, /*neg_b=*/false, b,
      /*c_mod=*/(short)0, c, /*reuse_a=*/false, /*reuse_b=*/false);
}

// ---------------------------------------------------------------------------
// DPP16 butterfly reductions over each 16-lane half (row ops stay in-row,
// so lanes 0-15 and 16-31 reduce independently). No LDS traffic, no waits.
// ---------------------------------------------------------------------------
template<int CTRL>
static __device__ __forceinline__ float dpp_mov(float x) {
  return __int_as_float(__builtin_amdgcn_update_dpp(
      __float_as_int(x), __float_as_int(x), CTRL, 0xF, 0xF, true));
}
static __device__ __forceinline__ float red_max16(float v) {
  v = fmaxf(v, dpp_mov<0xB1>(v));   // quad_perm [1,0,3,2]  (xor 1)
  v = fmaxf(v, dpp_mov<0x4E>(v));   // quad_perm [2,3,0,1]  (xor 2)
  v = fmaxf(v, dpp_mov<0x141>(v));  // ROW_HALF_MIRROR      (pairs across 4s)
  v = fmaxf(v, dpp_mov<0x140>(v));  // ROW_MIRROR           (pairs across 8s)
  return v;
}
static __device__ __forceinline__ float red_sum16(float v) {
  v += dpp_mov<0xB1>(v);
  v += dpp_mov<0x4E>(v);
  v += dpp_mov<0x141>(v);
  v += dpp_mov<0x140>(v);
  return v;
}

// ---------------------------------------------------------------------------
// GEMM: C[M,N] = A[M,K] (fp32 or bf16) * B[K,N] (fp32) + bias
//   mode 0: bf16 scatter into QKV workspace [3][32][16][512][64]
//   mode 1: fp32 store row-major [M,N]
// Block: 256 threads (8 waves). Tile 128x128x32. Each wave: 32x64 (2x4 WMMA).
// Software-pipelined: next tile's global loads issue before this tile's WMMAs.
// ---------------------------------------------------------------------------
#define ALD 48   // ushort stride (row = 96B, 16B multiple)
#define BLD 48

template<bool A_F32>
__global__ __launch_bounds__(256) void gemm_kernel(
    const void* __restrict__ Aptr, const float* __restrict__ B,
    const float* __restrict__ bias, int M, int N, int K,
    unsigned short* __restrict__ qkv_out, float* __restrict__ proj_out,
    int mode)
{
  __shared__ unsigned short As[128 * ALD];   // [128][ALD] row-major (M x K)
  __shared__ unsigned short Bst[128 * BLD];  // [128][BLD] (N x K, transposed)

  const int tid    = threadIdx.x;
  const int lane   = tid & 31;
  const int wave   = tid >> 5;
  const int lane15 = lane & 15;
  const int halfSel = lane >> 4;
  const int waveM  = (wave & 3) * 32;
  const int waveN  = (wave >> 2) * 64;
  const int rowBlk = blockIdx.y * 128;
  const int colBlk = blockIdx.x * 128;

  v8f acc[2][4];
  #pragma unroll
  for (int mt = 0; mt < 2; ++mt)
    #pragma unroll
    for (int nt = 0; nt < 4; ++nt) { v8f z = {}; acc[mt][nt] = z; }

  const int aRow = tid >> 1, aCol = (tid & 1) * 16;   // 2 thr/row, 16 cols each
  const int bRow = tid >> 3, bCol = (tid & 7) * 16;   // 8 thr/row, 16 cols each
  const size_t aGr = (size_t)(rowBlk + aRow);

  // pipeline registers holding the *next* tile's raw global data
  float4 aF[4]; uint4 aU[2]; float4 bF[4];

  // ---- prologue: load tile kb = 0 ----
  if (A_F32) {
    const float4* src = (const float4*)((const float*)Aptr + aGr * K + aCol);
    #pragma unroll
    for (int i = 0; i < 4; ++i) aF[i] = src[i];
  } else {
    const uint4* src = (const uint4*)((const unsigned short*)Aptr + aGr * K + aCol);
    aU[0] = src[0]; aU[1] = src[1];
  }
  {
    const float4* src = (const float4*)(B + (size_t)bRow * N + colBlk + bCol);
    #pragma unroll
    for (int i = 0; i < 4; ++i) bF[i] = src[i];
  }

  for (int kb = 0; kb < K; kb += 32) {
    __syncthreads();   // previous tile's compute finished; safe to overwrite LDS
    // ---- store staged registers to LDS (convert to bf16 if fp32) ----
    if (A_F32) {
      union { unsigned short us[16]; uint4 q[2]; } t;
      #pragma unroll
      for (int i = 0; i < 4; ++i) {
        t.us[i*4+0] = f2bf(aF[i].x); t.us[i*4+1] = f2bf(aF[i].y);
        t.us[i*4+2] = f2bf(aF[i].z); t.us[i*4+3] = f2bf(aF[i].w);
      }
      uint4* dst = (uint4*)&As[aRow * ALD + aCol];
      dst[0] = t.q[0]; dst[1] = t.q[1];
    } else {
      uint4* dst = (uint4*)&As[aRow * ALD + aCol];
      dst[0] = aU[0]; dst[1] = aU[1];
    }
    #pragma unroll
    for (int i = 0; i < 4; ++i) {
      Bst[(bCol + i*4 + 0) * BLD + bRow] = f2bf(bF[i].x);
      Bst[(bCol + i*4 + 1) * BLD + bRow] = f2bf(bF[i].y);
      Bst[(bCol + i*4 + 2) * BLD + bRow] = f2bf(bF[i].z);
      Bst[(bCol + i*4 + 3) * BLD + bRow] = f2bf(bF[i].w);
    }
    __syncthreads();

    // ---- issue next tile's global loads (overlap with WMMAs below) ----
    if (kb + 32 < K) {
      const int kn = kb + 32;
      if (A_F32) {
        const float4* src = (const float4*)((const float*)Aptr + aGr * K + kn + aCol);
        #pragma unroll
        for (int i = 0; i < 4; ++i) aF[i] = src[i];
      } else {
        const uint4* src = (const uint4*)((const unsigned short*)Aptr + aGr * K + kn + aCol);
        aU[0] = src[0]; aU[1] = src[1];
      }
      const float4* src = (const float4*)(B + (size_t)(kn + bRow) * N + colBlk + bCol);
      #pragma unroll
      for (int i = 0; i < 4; ++i) bF[i] = src[i];
    }

    // ---- fragments + WMMA ----
    v16bf a[2], b[4];
    const int kA = halfSel * 16;   // A: half-wave holds K 0..15 or 16..31
    #pragma unroll
    for (int mt = 0; mt < 2; ++mt) {
      const unsigned short* p = &As[(waveM + mt*16 + lane15) * ALD];
      a[mt] = make_frag(p + kA, p + kA + 8);
    }
    const int kB = halfSel * 16;   // B: half-wave holds contraction 0..15 or 16..31
    #pragma unroll
    for (int nt = 0; nt < 4; ++nt) {
      const unsigned short* p = &Bst[(waveN + nt*16 + lane15) * BLD];
      b[nt] = make_frag(p + kB, p + kB + 8);
    }
    #pragma unroll
    for (int mt = 0; mt < 2; ++mt)
      #pragma unroll
      for (int nt = 0; nt < 4; ++nt)
        acc[mt][nt] = wmma_bf16(a[mt], b[nt], acc[mt][nt]);
  }

  // ---- epilogue ----
  #pragma unroll
  for (int mt = 0; mt < 2; ++mt) {
    #pragma unroll
    for (int nt = 0; nt < 4; ++nt) {
      #pragma unroll
      for (int r = 0; r < 8; ++r) {
        const int row = rowBlk + waveM + mt*16 + r + halfSel*8;
        const int col = colBlk + waveN + nt*16 + lane15;
        const float v = acc[mt][nt][r] + bias[col];
        if (mode == 0) {
          const int which = col >> 10, h = (col >> 6) & 15, d = col & 63;
          const int win = row >> 9, wr = row & 511;
          const size_t idx = ((((size_t)which*32 + win)*16 + h)*512 + wr)*64 + d;
          qkv_out[idx] = f2bf(v);
        } else {
          proj_out[(size_t)row * N + col] = v;
        }
      }
    }
  }
}

// ---------------------------------------------------------------------------
// Flash attention per (window*head, 128-row Q chunk). 256 threads = 8 waves,
// each wave owns 16 Q rows. Causal within the window, scale = 1/sqrt(64).
// ---------------------------------------------------------------------------
#define KS_LD 72    // K rows:   64 + 8 pad  (144B rows, 16B multiple)
#define VT_LD 520   // V^T rows: 512 + 8 pad (1040B rows, 16B multiple)
#define PS_LD 48    // P rows:   32 + 16 pad (96B rows)

__global__ __launch_bounds__(256) void attn_kernel(
    const unsigned short* __restrict__ qkv, unsigned short* __restrict__ aout)
{
  extern __shared__ unsigned short lds[];
  unsigned short* Ks = lds;                    // [512][KS_LD]  (k-row major)
  unsigned short* Vt = Ks + 512 * KS_LD;       // [64][VT_LD]   (d-row major)
  unsigned short* Ps = Vt + 64 * VT_LD;        // [8 waves][16][PS_LD]

  const int tid = threadIdx.x, lane = tid & 31, wave = tid >> 5;
  const int lane15 = lane & 15, halfSel = lane >> 4;
  const int qblk = blockIdx.x;   // 0..3 (128-row Q chunks)
  const int wh   = blockIdx.y;   // win*16 + head, 0..511

  const size_t headElems = 512 * 64;
  const unsigned short* Qh = qkv + (size_t)(0*512 + wh) * headElems;
  const unsigned short* Kh = qkv + (size_t)(1*512 + wh) * headElems;
  const unsigned short* Vh = qkv + (size_t)(2*512 + wh) * headElems;

  const int kLimit = qblk * 128 + 128;   // causal: only rows < kLimit needed

  // ---- stage K via CDNA5 async global->LDS DMA (ASYNCcnt-tracked) ----
  // Dynamic LDS array starts at LDS offset 0 in this kernel, so byte offsets
  // into `lds` are valid per-lane LDS destination addresses.
  for (int i = tid; i < kLimit * 8; i += 256) {
    const int row = i >> 3, c8 = (i & 7) * 8;
    const unsigned ldsOff = (unsigned)(row * KS_LD + c8) * 2u;   // dest in Ks
    const unsigned glOff  = (unsigned)(row * 64 + c8) * 2u;      // src offset
    asm volatile("global_load_async_to_lds_b128 %0, %1, %2 offset:0"
                 :: "v"(ldsOff), "v"(glOff), "s"(Kh) : "memory");
  }
  // ---- stage V transposed: Vt[d][k] (manual: needs transpose) ----
  for (int i = tid; i < kLimit * 8; i += 256) {
    const int row = i >> 3, c8 = (i & 7) * 8;
    union { uint4 v; unsigned short e[8]; } t;
    t.v = *(const uint4*)(Vh + (size_t)row*64 + c8);
    #pragma unroll
    for (int d = 0; d < 8; ++d) Vt[(c8 + d) * VT_LD + row] = t.e[d];
  }
  asm volatile("s_wait_asynccnt 0" ::: "memory");  // this wave's DMAs done
  __syncthreads();                                 // all waves' staging visible

  const int qBase = qblk * 128 + wave * 16;

  // Q fragments (held in registers for the whole K loop): d 0..31 and 32..63
  v16bf qa[2];
  {
    const unsigned short* qp = Qh + (size_t)(qBase + lane15) * 64;
    const int kA = halfSel * 16;
    qa[0] = make_frag(qp + kA,      qp + kA + 8);
    qa[1] = make_frag(qp + 32 + kA, qp + 32 + kA + 8);
  }

  v8f o[4];
  #pragma unroll
  for (int dt = 0; dt < 4; ++dt) { v8f z = {}; o[dt] = z; }
  float mrow[8], lrow[8];
  #pragma unroll
  for (int r = 0; r < 8; ++r) { mrow[r] = -1e30f; lrow[r] = 0.f; }

  unsigned short* PsW = Ps + wave * 16 * PS_LD;

  for (int kt = 0; kt < qBase + 16; kt += 32) {
    // ---- scores: S[16 x 32] = Q @ K^T (4 WMMAs) ----
    v8f s0 = {}, s1 = {};
    {
      const unsigned short* kr0 = &Ks[(kt      + lane15) * KS_LD] + halfSel * 16;
      s0 = wmma_bf16(qa[0], make_frag(kr0,      kr0 + 8),  s0);
      s0 = wmma_bf16(qa[1], make_frag(kr0 + 32, kr0 + 40), s0);
      const unsigned short* kr1 = &Ks[(kt + 16 + lane15) * KS_LD] + halfSel * 16;
      s1 = wmma_bf16(qa[0], make_frag(kr1,      kr1 + 8),  s1);
      s1 = wmma_bf16(qa[1], make_frag(kr1 + 32, kr1 + 40), s1);
    }

    // ---- scale, causal mask, online softmax (DPP reductions, no LDS) ----
    float p0[8], p1[8];
    #pragma unroll
    for (int r = 0; r < 8; ++r) {
      const int qg = qBase + r + halfSel * 8;
      float a = s0[r] * 0.125f;
      float b = s1[r] * 0.125f;
      if (kt      + lane15 > qg) a = -1e9f;
      if (kt + 16 + lane15 > qg) b = -1e9f;
      const float mx    = red_max16(fmaxf(a, b));
      const float nm    = fmaxf(mrow[r], mx);
      const float alpha = __expf(mrow[r] - nm);
      const float e0 = __expf(a - nm), e1 = __expf(b - nm);
      const float rs = red_sum16(e0 + e1);
      lrow[r] = lrow[r] * alpha + rs;
      mrow[r] = nm;
      p0[r] = e0; p1[r] = e1;
      #pragma unroll
      for (int dt = 0; dt < 4; ++dt) o[dt][r] *= alpha;
    }

    // ---- re-layout P (C/D layout -> A layout) through per-wave LDS ----
    #pragma unroll
    for (int r = 0; r < 8; ++r) {
      const int m = r + halfSel * 8;
      PsW[m * PS_LD + lane15]      = f2bf(p0[r]);
      PsW[m * PS_LD + 16 + lane15] = f2bf(p1[r]);
    }
    asm volatile("s_wait_dscnt 0" ::: "memory");   // wave-local LDS RAW fence
    const unsigned short* pb = &PsW[lane15 * PS_LD] + halfSel * 16;
    const v16bf pf = make_frag(pb, pb + 8);

    // ---- O += P @ V (4 WMMAs, V fragments contiguous from Vt) ----
    #pragma unroll
    for (int dt = 0; dt < 4; ++dt) {
      const unsigned short* vb = &Vt[(dt*16 + lane15) * VT_LD + kt] + halfSel * 16;
      o[dt] = wmma_bf16(pf, make_frag(vb, vb + 8), o[dt]);
    }
  }

  // ---- normalize and write attn output: [16384][1024] bf16, col = h*64+d ----
  const int win = wh >> 4, h = wh & 15;
  #pragma unroll
  for (int r = 0; r < 8; ++r) {
    const float inv = 1.0f / lrow[r];
    const int row = qBase + r + halfSel * 8;
    const size_t base = ((size_t)win * 512 + row) * 1024 + h * 64;
    #pragma unroll
    for (int dt = 0; dt < 4; ++dt)
      aout[base + dt*16 + lane15] = f2bf(o[dt][r] * inv);
  }
}

// ---------------------------------------------------------------------------
// Launch: QKV GEMM -> attention -> output projection
// ---------------------------------------------------------------------------
extern "C" void kernel_launch(void* const* d_in, const int* in_sizes, int n_in,
                              void* d_out, int out_size, void* d_ws, size_t ws_size,
                              hipStream_t stream) {
  const float* x     = (const float*)d_in[0];   // [4,4096,1024]
  const float* w_qkv = (const float*)d_in[1];   // [1024,3072]
  const float* b_qkv = (const float*)d_in[2];   // [3072]
  const float* w_o   = (const float*)d_in[3];   // [1024,1024]
  const float* b_o   = (const float*)d_in[4];   // [1024]
  float* out = (float*)d_out;                   // [4,4096,1024]

  // workspace: QKV bf16 [3][32][16][512][64] then attn-out bf16 [16384][1024]
  unsigned short* qkv_ws  = (unsigned short*)d_ws;
  unsigned short* attn_ws = qkv_ws + (size_t)3 * 512 * 512 * 64;

  dim3 blk(256);

  // 1) QKV projection: [16384,1024] x [1024,3072]
  gemm_kernel<true><<<dim3(3072/128, 16384/128), blk, 0, stream>>>(
      x, w_qkv, b_qkv, 16384, 3072, 1024, qkv_ws, nullptr, 0);

  // 2) attention: one block per (win*head, 128-row Q chunk)
  const size_t ldsBytes =
      (size_t)(512 * KS_LD + 64 * VT_LD + 8 * 16 * PS_LD) * sizeof(unsigned short);
  attn_kernel<<<dim3(4, 512), blk, ldsBytes, stream>>>(qkv_ws, attn_ws);

  // 3) output projection: [16384,1024](bf16) x [1024,1024] -> fp32 + bias
  gemm_kernel<false><<<dim3(1024/128, 16384/128), blk, 0, stream>>>(
      attn_ws, w_o, b_o, 16384, 1024, 1024, nullptr, out, 1);
}
